// NTM_7000796692715
// MI455X (gfx1250) — matrix operand order repaired and checked by reading, not compiled, not run on hardware
//
#include <hip/hip_runtime.h>

// ---------------------------------------------------------------------------
// NTM forward:  N=32, T=64, C1=128, C2=256, H=512
// d_out = [ hs (32,64,512) | cs (32,64,128,256) ]  (float32)
// 6 fused stages per timestep (min serial depth of the dependency DAG).
// ---------------------------------------------------------------------------

#define NB   32
#define TT   64
#define C1   128
#define C2   256
#define HH   512
#define G3   (3 * HH)          // 1536

typedef _Float16 v16h __attribute__((ext_vector_type(16)));
typedef float    v8f  __attribute__((ext_vector_type(8)));

__device__ __forceinline__ float dot4(float4 a, float4 b) {
  return a.x * b.x + a.y * b.y + a.z * b.z + a.w * b.w;
}
__device__ __forceinline__ float sigmoidf_(float x) {
  return 1.0f / (1.0f + __expf(-x));
}

// ---------------------------------------------------------------------------
// WMMA 16x16 tile body.  Fragment layouts per CDNA5 ISA (wave32):
//   A (16x32 f16): lanes 0-15 -> M, b=lane>>4; slots 0-7 = K[b*8 .. b*8+7],
//                  slots 8-15 = K[16+b*8 .. 16+b*8+7]
//   B (32x16 f16): lanes 0-15 -> N, same K slot order as A
//   C/D (16x16 f32): lane n = l&15, row m = (l>>4)*8 + r  (r = VGPR index)
// ---------------------------------------------------------------------------
__device__ __forceinline__ void wmma_tile(
    float* __restrict__ out, const _Float16* __restrict__ A,
    const _Float16* __restrict__ W, const float* __restrict__ bias,
    int N, int K, int act, int n0, int m0, int lane) {
  const int b  = lane >> 4;
  const int lm = lane & 15;
  v8f acc = {};
  for (int k0 = 0; k0 < K; k0 += 32) {
    v16h af, bf;
    const _Float16* Arow = A + (long)(m0 + lm) * K + k0 + b * 8;
#pragma unroll
    for (int s = 0; s < 8; ++s) af[s] = Arow[s];
#pragma unroll
    for (int s = 0; s < 8; ++s) af[8 + s] = Arow[16 + s];
    const _Float16* Wrow = W + (long)(n0 + lm) * K + k0 + b * 8;
#pragma unroll
    for (int s = 0; s < 8; ++s) bf[s] = Wrow[s];
#pragma unroll
    for (int s = 0; s < 8; ++s) bf[8 + s] = Wrow[16 + s];
    acc = __builtin_amdgcn_wmma_f32_16x16x32_f16(false, af, false, bf,
                                                 (short)0, acc, false, false);
  }
  const int n     = n0 + lm;
  const int mbase = m0 + b * 8;
  const float bv  = bias[n];
#pragma unroll
  for (int r = 0; r < 8; ++r) {
    float v = acc[r] + bv;
    if (act == 1) v = sigmoidf_(v);
    out[(long)(mbase + r) * N + n] = v;
  }
}

// ---------------------------------------------------------------------------
// f32 -> f16 conversion (weights once) and h init
// ---------------------------------------------------------------------------
__global__ void f32_to_f16_kernel(_Float16* __restrict__ dst,
                                  const float* __restrict__ src, int n) {
  int i = blockIdx.x * 256 + threadIdx.x;
  if (i < n) dst[i] = (_Float16)src[i];
}

__global__ void init_h_kernel(float* __restrict__ h, _Float16* __restrict__ h16,
                              const float* __restrict__ h0, int n) {
  int i = blockIdx.x * 256 + threadIdx.x;
  if (i < n) { float v = h0[i]; h[i] = v; h16[i] = (_Float16)v; }
}

// ---------------------------------------------------------------------------
// Stage 1 "head": k = h@Wk^T+bk (WMMA), gh = h@W_hh^T+b_hh (WMMA), beta (GEMV).
// All read h_prev only -> one fused kernel. Routing is block-uniform, so the
// WMMA blocks keep EXEC all-ones.
// grid = (C2/16 + G3/16 + 1, 2), block = 32 (one wave).
// ---------------------------------------------------------------------------
__global__ __launch_bounds__(32) void head_kernel(
    const _Float16* __restrict__ h16, const float* __restrict__ hf,
    const _Float16* __restrict__ Wk16, const float* __restrict__ bk,
    const _Float16* __restrict__ Whh16, const float* __restrict__ bhh,
    const float* __restrict__ Wb, const float* __restrict__ bb,
    float* __restrict__ kout, float* __restrict__ ghout,
    float* __restrict__ betaout) {
  const int bx   = blockIdx.x;
  const int lane = threadIdx.x;
  const int nk   = C2 / 16;            // 16 tiles for k
  const int ngh  = G3 / 16;            // 96 tiles for gh

  if (bx == nk + ngh) {                // beta block
    if (blockIdx.y != 0) return;
    int n = lane;                      // 0..31
    float acc = bb[0];
    for (int j = 0; j < HH; ++j) acc += hf[n * HH + j] * Wb[j];
    float bpos = fmaxf(acc, 0.0f);
    float bneg = fminf(acc, 0.0f);
    betaout[n] = log1pf(__expf(bneg)) + bpos + log1pf(__expf(-bpos))
               + (1.0f - 0.6931471805599453f);
    return;
  }
  const int m0 = blockIdx.y * 16;
  if (bx < nk) {
    wmma_tile(kout, h16, Wk16, bk, C2, HH, 0, bx * 16, m0, lane);
  } else {
    wmma_tile(ghout, h16, Whh16, bhh, G3, HH, 0, (bx - nk) * 16, m0, lane);
  }
}

// ---------------------------------------------------------------------------
// Stage 3: gi = r @ W_ih^T + b_ih   (WMMA, N=1536, K=256)
// ---------------------------------------------------------------------------
__global__ __launch_bounds__(32) void gi_kernel(
    float* __restrict__ out, const _Float16* __restrict__ r16,
    const _Float16* __restrict__ Wih16, const float* __restrict__ bih) {
  wmma_tile(out, r16, Wih16, bih, G3, C2, 0,
            blockIdx.x * 16, blockIdx.y * 16, threadIdx.x);
}

// ---------------------------------------------------------------------------
// Stage 5: e = sigmoid(h@We^T+be), v = h@Wv^T+bv, fused via blockIdx.z.
// grid = (C2/16, 2, 2), block = 32.
// ---------------------------------------------------------------------------
__global__ __launch_bounds__(32) void ev_kernel(
    const _Float16* __restrict__ h16,
    const _Float16* __restrict__ We16, const float* __restrict__ be,
    const _Float16* __restrict__ Wv16, const float* __restrict__ bv,
    float* __restrict__ eout, float* __restrict__ vout) {
  const int n0 = blockIdx.x * 16;
  const int m0 = blockIdx.y * 16;
  if (blockIdx.z == 0)
    wmma_tile(eout, h16, We16, be, C2, HH, 1, n0, m0, threadIdx.x);
  else
    wmma_tile(vout, h16, Wv16, bv, C2, HH, 0, n0, m0, threadIdx.x);
}

// ---------------------------------------------------------------------------
// Stage 2: cosine attention + softmax + weighted read. One block per batch n.
//   num[i] = cx[n,i,:] . k[n,:];  cn[i] = ||cx[n,i,:]||;  kn = ||k[n,:]||
//   s = num / max(cn*kn, eps);  w = softmax(s*beta);  r[c] = sum_i w[i]*cx[n,i,c]
// ---------------------------------------------------------------------------
__global__ __launch_bounds__(256) void attn_kernel(
    const float* __restrict__ cx,     // c_x + t*C1*C2 ; batch stride = T*C1*C2
    const float* __restrict__ kvec,   // (N, C2)
    const float* __restrict__ beta,   // (N)
    float* __restrict__ w,            // (N, C1)
    float* __restrict__ r,            // (N, C2)
    _Float16* __restrict__ r16) {
  __shared__ float s_s[C1];
  const int n    = blockIdx.x;
  const int tid  = threadIdx.x;
  const int wv   = tid >> 5;
  const int lane = tid & 31;
  const float* cxn = cx + (long)n * (TT * C1 * C2);
  const float  bet = beta[n];

  // k fragments (8 floats per lane) + k norm (redundant per wave, cheap)
  const float* kn_ptr = kvec + n * C2;
  float4 ka = *(const float4*)(kn_ptr + lane * 8);
  float4 kb = *(const float4*)(kn_ptr + lane * 8 + 4);
  float kn2 = dot4(ka, ka) + dot4(kb, kb);
#pragma unroll
  for (int off = 16; off; off >>= 1) kn2 += __shfl_xor(kn2, off);
  const float knorm = sqrtf(kn2);

  // pass 1: similarity scores (one wave per memory row i)
  for (int i = wv; i < C1; i += 8) {
    const float* row = cxn + (long)i * C2;
    float4 xa = *(const float4*)(row + lane * 8);
    float4 xb = *(const float4*)(row + lane * 8 + 4);
    float num = dot4(xa, ka) + dot4(xb, kb);
    float cn2 = dot4(xa, xa) + dot4(xb, xb);
#pragma unroll
    for (int off = 16; off; off >>= 1) {
      num += __shfl_xor(num, off);
      cn2 += __shfl_xor(cn2, off);
    }
    if (lane == 0) {
      float s = num / fmaxf(sqrtf(cn2) * knorm, 1e-8f);
      s_s[i] = s * bet;
    }
  }
  __syncthreads();

  // softmax over 128 entries: wave 0 handles 4 values per lane
  if (wv == 0) {
    float v0 = s_s[lane], v1 = s_s[lane + 32];
    float v2 = s_s[lane + 64], v3 = s_s[lane + 96];
    float mx = fmaxf(fmaxf(v0, v1), fmaxf(v2, v3));
#pragma unroll
    for (int off = 16; off; off >>= 1) mx = fmaxf(mx, __shfl_xor(mx, off));
    float e0 = __expf(v0 - mx), e1 = __expf(v1 - mx);
    float e2 = __expf(v2 - mx), e3 = __expf(v3 - mx);
    float sum = e0 + e1 + e2 + e3;
#pragma unroll
    for (int off = 16; off; off >>= 1) sum += __shfl_xor(sum, off);
    float inv = 1.0f / sum;
    s_s[lane]      = e0 * inv;  s_s[lane + 32] = e1 * inv;
    s_s[lane + 64] = e2 * inv;  s_s[lane + 96] = e3 * inv;
    float* wn = w + n * C1;
    wn[lane]      = e0 * inv;  wn[lane + 32] = e1 * inv;
    wn[lane + 64] = e2 * inv;  wn[lane + 96] = e3 * inv;
  }
  __syncthreads();

  // pass 2: weighted read r[c] (coalesced column gather; cx slice hits L2)
  const int c = tid;            // 0..255
  float acc = 0.0f;
  for (int i = 0; i < C1; ++i) acc += s_s[i] * cxn[(long)i * C2 + c];
  r[n * C2 + c]   = acc;
  r16[n * C2 + c] = (_Float16)acc;
}

// ---------------------------------------------------------------------------
// Stage 4: GRU combine; writes h state, h16, and hs[:,t,:]
// ---------------------------------------------------------------------------
__global__ __launch_bounds__(256) void gru_kernel(
    const float* __restrict__ gi, const float* __restrict__ gh,
    float* __restrict__ h, _Float16* __restrict__ h16,
    float* __restrict__ hs, int t) {
  int idx = blockIdx.x * 256 + threadIdx.x;  // 0 .. N*H-1
  int n = idx >> 9, j = idx & (HH - 1);
  const float* gin = gi + n * G3;
  const float* ghn = gh + n * G3;
  float rg = sigmoidf_(gin[j] + ghn[j]);
  float z  = sigmoidf_(gin[HH + j] + ghn[HH + j]);
  float nn = tanhf(gin[2 * HH + j] + rg * ghn[2 * HH + j]);
  float hp = h[idx];
  float hv = (1.0f - z) * nn + z * hp;
  h[idx]   = hv;
  h16[idx] = (_Float16)hv;
  hs[(long)n * (TT * HH) + (long)t * HH + j] = hv;
}

// ---------------------------------------------------------------------------
// Stage 6: memory update  c_new = c_prev*(1 - w*e) + w*v  (float4 streaming,
// the HBM-roofline stage: ~8 MB/step)
// ---------------------------------------------------------------------------
__global__ __launch_bounds__(256) void cupd_kernel(
    const float* __restrict__ cprev, const float* __restrict__ w,
    const float* __restrict__ e, const float* __restrict__ v,
    float* __restrict__ cout, long cp_bstride, long co_bstride) {
  long idx = (long)blockIdx.x * 256 + threadIdx.x;   // float4 index
  long f   = idx * 4;
  int n   = (int)(f >> 15);        // / (C1*C2)
  int rem = (int)(f & 32767);
  int i   = rem >> 8;
  int c   = rem & 255;
  float wi  = w[n * C1 + i];
  float4 ev = *(const float4*)(e + n * C2 + c);
  float4 vv = *(const float4*)(v + n * C2 + c);
  const float* cpp = cprev + (long)n * cp_bstride + (long)i * C2 + c;
  __builtin_prefetch(cpp + C2, 0, 1);   // global_prefetch_b8: stream-ahead
  float4 cp = *(const float4*)cpp;
  float4 o;
  o.x = cp.x * (1.0f - wi * ev.x) + wi * vv.x;
  o.y = cp.y * (1.0f - wi * ev.y) + wi * vv.y;
  o.z = cp.z * (1.0f - wi * ev.z) + wi * vv.z;
  o.w = cp.w * (1.0f - wi * ev.w) + wi * vv.w;
  *(float4*)(cout + (long)n * co_bstride + (long)i * C2 + c) = o;
}

// ---------------------------------------------------------------------------
extern "C" void kernel_launch(void* const* d_in, const int* in_sizes, int n_in,
                              void* d_out, int out_size, void* d_ws, size_t ws_size,
                              hipStream_t stream) {
  const float* h0   = (const float*)d_in[0];
  const float* c0   = (const float*)d_in[1];
  const float* c_x  = (const float*)d_in[2];
  const float* Wk   = (const float*)d_in[3];
  const float* bk   = (const float*)d_in[4];
  const float* Wb   = (const float*)d_in[5];
  const float* bb   = (const float*)d_in[6];
  const float* We   = (const float*)d_in[7];
  const float* be   = (const float*)d_in[8];
  const float* Wv   = (const float*)d_in[9];
  const float* bv   = (const float*)d_in[10];
  const float* W_ih = (const float*)d_in[11];
  const float* b_ih = (const float*)d_in[12];
  const float* W_hh = (const float*)d_in[13];
  const float* b_hh = (const float*)d_in[14];

  float* hs = (float*)d_out;                         // (N,T,H)
  float* cs = hs + (long)NB * TT * HH;               // (N,T,C1,C2)

  // workspace bump allocator
  char* ws = (char*)d_ws;
  auto alloc = [&](size_t bytes) -> void* {
    void* p = (void*)ws;
    ws += (bytes + 255) & ~(size_t)255;
    return p;
  };
  _Float16* Wk16  = (_Float16*)alloc((size_t)C2 * HH * 2);
  _Float16* We16  = (_Float16*)alloc((size_t)C2 * HH * 2);
  _Float16* Wv16  = (_Float16*)alloc((size_t)C2 * HH * 2);
  _Float16* Wih16 = (_Float16*)alloc((size_t)G3 * C2 * 2);
  _Float16* Whh16 = (_Float16*)alloc((size_t)G3 * HH * 2);
  float*    hbuf  = (float*)alloc((size_t)NB * HH * 4);
  _Float16* h16   = (_Float16*)alloc((size_t)NB * HH * 2);
  float*    kbuf  = (float*)alloc((size_t)NB * C2 * 4);
  float*    betab = (float*)alloc((size_t)NB * 4);
  float*    ghbuf = (float*)alloc((size_t)NB * G3 * 4);
  float*    gibuf = (float*)alloc((size_t)NB * G3 * 4);
  float*    wbuf  = (float*)alloc((size_t)NB * C1 * 4);
  float*    rbuf  = (float*)alloc((size_t)NB * C2 * 4);
  _Float16* r16   = (_Float16*)alloc((size_t)NB * C2 * 2);
  float*    ebuf  = (float*)alloc((size_t)NB * C2 * 4);
  float*    vbuf  = (float*)alloc((size_t)NB * C2 * 4);

  // one-time (per call) weight conversion + h init
  auto conv = [&](_Float16* dst, const float* src, int n) {
    f32_to_f16_kernel<<<(n + 255) / 256, 256, 0, stream>>>(dst, src, n);
  };
  conv(Wk16, Wk, C2 * HH);
  conv(We16, We, C2 * HH);
  conv(Wv16, Wv, C2 * HH);
  conv(Wih16, W_ih, G3 * C2);
  conv(Whh16, W_hh, G3 * HH);
  init_h_kernel<<<(NB * HH + 255) / 256, 256, 0, stream>>>(hbuf, h16, h0, NB * HH);

  const long cs_bstride = (long)TT * C1 * C2;   // batch stride inside cs
  const long c0_bstride = (long)C1 * C2;

  for (int t = 0; t < TT; ++t) {
    // 1) head: k-GEMM + gh-GEMM + beta (all read h_prev)
    head_kernel<<<dim3(C2 / 16 + G3 / 16 + 1, 2), 32, 0, stream>>>(
        h16, hbuf, Wk16, bk, Whh16, b_hh, Wb, bb, kbuf, ghbuf, betab);
    // 2) attention -> w, r
    attn_kernel<<<NB, 256, 0, stream>>>(
        c_x + (long)t * C1 * C2, kbuf, betab, wbuf, rbuf, r16);
    // 3) gi = r @ W_ih^T + b_ih
    gi_kernel<<<dim3(G3 / 16, 2), 32, 0, stream>>>(gibuf, r16, Wih16, b_ih);
    // 4) GRU -> h_new, hs[:,t,:]
    gru_kernel<<<(NB * HH) / 256, 256, 0, stream>>>(
        gibuf, ghbuf, hbuf, h16, hs, t);
    // 5) e = sigmoid(h@We^T+be) and v = h@Wv^T+bv (fused)
    ev_kernel<<<dim3(C2 / 16, 2, 2), 32, 0, stream>>>(
        h16, We16, be, Wv16, bv, ebuf, vbuf);
    // 6) c update -> cs[:,t,:,:]
    const float* cprev = (t == 0) ? c0 : (cs + (long)(t - 1) * C1 * C2);
    long cpstride = (t == 0) ? c0_bstride : cs_bstride;
    cupd_kernel<<<(NB * C1 * C2 / 4) / 256, 256, 0, stream>>>(
        cprev, wbuf, ebuf, vbuf, cs + (long)t * C1 * C2, cpstride, cs_bstride);
  }
}
